// SVDHead_Mask_67791763800493
// MI455X (gfx1250) — matrix operand order repaired and checked by reading, not compiled, not run on hardware
//
#include <hip/hip_runtime.h>
#include <hip/hip_bf16.h>
#include <math.h>

// ---------------- problem constants ----------------
#define Bb 16
#define Dd 512
#define Nn 2048
#define Mm 2048

// ---------------- kernel-1 tiling ------------------
#define ROWS    128          // N-rows per block
#define THREADS 256          // 8 wave32
#define MT      16           // M columns per streamed tile
#define AS      (Dd + 8)     // padded LDS row stride (elements) -> 16B align kept, bank conflicts killed
#define BS      (Dd + 8)
#define RSCALE  0.04419417382415922f   // 1/sqrt(512)

typedef __attribute__((ext_vector_type(16))) __bf16 v16bf;
typedef __attribute__((ext_vector_type(8)))  __bf16 v8bf;
typedef __attribute__((ext_vector_type(8)))  float  v8f;

// =====================================================================
// Kernel 1: fused scores GEMM (bf16 WMMA) + online softmax + 5-row V
// products.  Block = 8 waves, each wave owns a 16-row strip; the block
// streams tgt_embedding in 16-column tiles with register double-buffered
// staging so L2 latency overlaps the WMMA loop.
// =====================================================================
__global__ __launch_bounds__(THREADS)
void flash_corr_kernel(const float* __restrict__ src_emb,   // [B][D][N]
                       const float* __restrict__ tgt_emb,   // [B][D][M]
                       const float* __restrict__ tgt,       // [B][3][M]
                       const float* __restrict__ tgt_dmask, // [B][1][M]
                       const float* __restrict__ tgt_imask, // [B][1][M]
                       const float* __restrict__ src_dmask, // [B][1][N]
                       const float* __restrict__ src_imask, // [B][1][N]
                       float* __restrict__ ws_corr,         // [B][3][N]
                       float* __restrict__ ws_mval)         // [B][N]
{
    extern __shared__ char smem[];
    __bf16* Abuf = (__bf16*)smem;                               // [ROWS][AS]
    __bf16* Bbuf = (__bf16*)(smem + (size_t)ROWS * AS * 2);     // [MT][BS]
    float*  Vbuf = (float*)(smem + (size_t)(ROWS * AS + MT * BS) * 2); // [5][MT]

    const int tid   = threadIdx.x;
    const int blk   = blockIdx.x;
    const int b     = blk / (Nn / ROWS);
    const int n0    = (blk % (Nn / ROWS)) * ROWS;

    const float* A  = src_emb + (size_t)b * Dd * Nn;   // A[d][n]
    const float* Bm = tgt_emb + (size_t)b * Dd * Mm;   // B[d][m]

    // ---- stage A tile once: LDS[n_local][d] (bf16) ----
    for (int i = tid; i < ROWS * Dd; i += THREADS) {
        int nl = i % ROWS;                  // contiguous global reads across tid
        int d  = i / ROWS;
        Abuf[(size_t)nl * AS + d] = (__bf16)A[(size_t)d * Nn + n0 + nl];
    }

    const int lane = tid & 31;
    const int wave = tid >> 5;
    const int hk   = lane >> 4;     // half-wave id
    const int lcol = lane & 15;     // column within 16x16 tile
    const int rowbase = wave * 16;

    // ---- staging registers (double buffer of the B/V tile) ----
    // 256 threads, 16x512 tile: each thread owns column scol = tid&15 and
    // d = sd0 + 16*u, u=0..31.  Column is constant per thread.
    const int scol = tid & 15;
    const int sd0  = tid >> 4;                         // 0..15
    const float* bptr = Bm + (size_t)sd0 * Mm + scol;  // element (d=sd0, m=scol)
    const int vj = tid >> 4, vc = tid & 15;            // V stager ids (tid < 80)

    float breg[32];
    float vreg = 0.f;
    {   // prologue: fetch tile 0
#pragma unroll
        for (int u = 0; u < 32; ++u)
            breg[u] = bptr[(size_t)u * 16 * Mm + 0];
        if (tid < 5 * MT) {
            if (vj < 3)       vreg = tgt[((size_t)b * 3 + vj) * Mm + vc];
            else if (vj == 3) vreg = tgt_dmask[(size_t)b * Mm + vc];
            else              vreg = tgt_imask[(size_t)b * Mm + vc];
        }
    }

    // per-lane online-softmax state: 8 rows x (max, sum, 5 value accumulators)
    float run_max[8], run_sum[8], accv[8][5];
#pragma unroll
    for (int i = 0; i < 8; ++i) {
        run_max[i] = -1e30f;
        run_sum[i] = 0.f;
#pragma unroll
        for (int j = 0; j < 5; ++j) accv[i][j] = 0.f;
    }

    for (int mt = 0; mt < Mm / MT; ++mt) {
        __syncthreads();   // all waves done reading the previous LDS tile

        // ---- commit staged tile mt to LDS (convert f32 -> bf16) ----
#pragma unroll
        for (int u = 0; u < 32; ++u)
            Bbuf[(size_t)scol * BS + sd0 + u * 16] = (__bf16)breg[u];
        if (tid < 5 * MT) Vbuf[vj * MT + vc] = vreg;

        // ---- prefetch tile mt+1 into registers (overlaps compute below) ----
        if (mt + 1 < Mm / MT) {
            const int m1 = (mt + 1) * MT;
#pragma unroll
            for (int u = 0; u < 32; ++u)
                breg[u] = bptr[(size_t)u * 16 * Mm + m1];
            if (tid < 5 * MT) {
                if (vj < 3)       vreg = tgt[((size_t)b * 3 + vj) * Mm + m1 + vc];
                else if (vj == 3) vreg = tgt_dmask[(size_t)b * Mm + m1 + vc];
                else              vreg = tgt_imask[(size_t)b * Mm + m1 + vc];
            }
        }
        __syncthreads();   // LDS tile mt visible

        // hoist this lane's V column
        float vv[5];
#pragma unroll
        for (int j = 0; j < 5; ++j) vv[j] = Vbuf[j * MT + lcol];

        // ---- 16x16 score tile via 16 x WMMA bf16 (K=32 each) ----
        v8f acc = {};
#pragma unroll
        for (int k = 0; k < Dd / 32; ++k) {
            const int k0 = k * 32;
            // A fragment (16x32): lanes 0-15 -> K {0..7,16..23}; lanes 16-31 -> K {8..15,24..31}
            const __bf16* ap = &Abuf[(size_t)(rowbase + lcol) * AS + k0 + hk * 8];
            v8bf alo = *(const v8bf*)ap;
            v8bf ahi = *(const v8bf*)(ap + 16);
            v16bf af;
#pragma unroll
            for (int e = 0; e < 8; ++e) { af[e] = alo[e]; af[e + 8] = ahi[e]; }
            // B fragment (32x16): lane col = lcol, lanes 0-15 -> K {0..7,16..23}; lanes 16-31 -> K {8..15,24..31}
            const __bf16* bp = &Bbuf[(size_t)lcol * BS + k0 + hk * 8];
            v8bf blo = *(const v8bf*)bp;
            v8bf bhi = *(const v8bf*)(bp + 16);
            v16bf bfv;
#pragma unroll
            for (int e = 0; e < 8; ++e) { bfv[e] = blo[e]; bfv[e + 8] = bhi[e]; }
            acc = __builtin_amdgcn_wmma_f32_16x16x32_bf16(
                      false, af, false, bfv, (short)0, acc, false, false);
        }

        // ---- online softmax update (rows i, my column lcol) ----
#pragma unroll
        for (int i = 0; i < 8; ++i) {
            float s = acc[i] * RSCALE;
            float tmax = s;
            tmax = fmaxf(tmax, __shfl_xor(tmax, 1));
            tmax = fmaxf(tmax, __shfl_xor(tmax, 2));
            tmax = fmaxf(tmax, __shfl_xor(tmax, 4));
            tmax = fmaxf(tmax, __shfl_xor(tmax, 8));
            float nm   = fmaxf(run_max[i], tmax);
            float corr = __expf(run_max[i] - nm);
            float p    = __expf(s - nm);
            run_max[i] = nm;
            run_sum[i] = run_sum[i] * corr + p;
#pragma unroll
            for (int j = 0; j < 5; ++j)
                accv[i][j] = accv[i][j] * corr + p * vv[j];
        }
    }

    // ---- finalize: reduce across the 16-lane half-wave, write results ----
#pragma unroll
    for (int i = 0; i < 8; ++i) {
        float ssum = run_sum[i];
        ssum += __shfl_xor(ssum, 1); ssum += __shfl_xor(ssum, 2);
        ssum += __shfl_xor(ssum, 4); ssum += __shfl_xor(ssum, 8);
        float inv = 1.0f / ssum;
        float o[5];
#pragma unroll
        for (int j = 0; j < 5; ++j) {
            float v = accv[i][j];
            v += __shfl_xor(v, 1); v += __shfl_xor(v, 2);
            v += __shfl_xor(v, 4); v += __shfl_xor(v, 8);
            o[j] = v * inv;
        }
        if (lcol == 0) {
            // C layout: lanes 0-15 hold rows i, lanes 16-31 hold rows i+8
            int n = n0 + rowbase + i + hk * 8;
            ws_corr[((size_t)b * 3 + 0) * Nn + n] = o[0];
            ws_corr[((size_t)b * 3 + 1) * Nn + n] = o[1];
            ws_corr[((size_t)b * 3 + 2) * Nn + n] = o[2];
            float depth = 0.5f * (src_dmask[(size_t)b * Nn + n] + o[3]);
            float img   = 0.5f * (src_imask[(size_t)b * Nn + n] + o[4]);
            float mval  = (1.f / (1.f + __expf(-depth))) * (1.f / (1.f + __expf(-img)));
            ws_mval[(size_t)b * Nn + n] = mval;
        }
    }
}

// =====================================================================
// Kernel 2: per-batch reductions + masked 3x3 covariance + 3x3 SVD
// =====================================================================
enum { OP_ADD = 0, OP_MIN = 1, OP_MAX = 2 };

__device__ __forceinline__ float block_red(float v, float* red, int op)
{
#pragma unroll
    for (int m = 1; m < 32; m <<= 1) {
        float o = __shfl_xor(v, m);
        v = (op == OP_ADD) ? v + o : (op == OP_MIN ? fminf(v, o) : fmaxf(v, o));
    }
    const int lane = threadIdx.x & 31, w = threadIdx.x >> 5;
    if (lane == 0) red[w] = v;
    __syncthreads();
    if (threadIdx.x == 0) {
        float r = red[0];
        for (int i = 1; i < 8; ++i) {
            float o = red[i];
            r = (op == OP_ADD) ? r + o : (op == OP_MIN ? fminf(r, o) : fmaxf(r, o));
        }
        red[8] = r;
    }
    __syncthreads();
    float r = red[8];
    __syncthreads();
    return r;
}

__global__ __launch_bounds__(256)
void reduce_svd_kernel(const float* __restrict__ src,     // [B][3][N]
                       const float* __restrict__ ws_corr, // [B][3][N]
                       const float* __restrict__ ws_mval, // [B][N]
                       float* __restrict__ out)           // R[16*9] ++ t[16*3]
{
    __shared__ float red[16];
    const int b = blockIdx.x, tid = threadIdx.x;
    const float* srcb = src     + (size_t)b * 3 * Nn;
    const float* corb = ws_corr + (size_t)b * 3 * Nn;
    const float* mvb  = ws_mval + (size_t)b * Nn;

    // ---- pass 1: min/max of mask + means of src and src_corr ----
    float lmin = 1e30f, lmax = -1e30f;
    float ss0 = 0, ss1 = 0, ss2 = 0, cs0 = 0, cs1 = 0, cs2 = 0;
    for (int n = tid; n < Nn; n += 256) {
        float m = mvb[n];
        lmin = fminf(lmin, m); lmax = fmaxf(lmax, m);
        ss0 += srcb[n]; ss1 += srcb[Nn + n]; ss2 += srcb[2 * Nn + n];
        cs0 += corb[n]; cs1 += corb[Nn + n]; cs2 += corb[2 * Nn + n];
    }
    float mmin = block_red(lmin, red, OP_MIN);
    float mmax = block_red(lmax, red, OP_MAX);
    const float invN = 1.0f / (float)Nn;
    float sm0 = block_red(ss0, red, OP_ADD) * invN;
    float sm1 = block_red(ss1, red, OP_ADD) * invN;
    float sm2 = block_red(ss2, red, OP_ADD) * invN;
    float cm0 = block_red(cs0, red, OP_ADD) * invN;
    float cm1 = block_red(cs1, red, OP_ADD) * invN;
    float cm2 = block_red(cs2, red, OP_ADD) * invN;
    float inv_range = 1.0f / fmaxf(mmax - mmin, 1e-6f);

    // ---- pass 2: H[i][j] = sum_n (src_i - sm_i) * w_n * (corr_j - cm_j) ----
    float h[9];
#pragma unroll
    for (int k = 0; k < 9; ++k) h[k] = 0.f;
    for (int n = tid; n < Nn; n += 256) {
        float w  = (mvb[n] - mmin) * inv_range;
        float a0 = srcb[n]           - sm0;
        float a1 = srcb[Nn + n]      - sm1;
        float a2 = srcb[2 * Nn + n]  - sm2;
        float c0 = (corb[n]          - cm0) * w;
        float c1 = (corb[Nn + n]     - cm1) * w;
        float c2 = (corb[2 * Nn + n] - cm2) * w;
        h[0] += a0 * c0; h[1] += a0 * c1; h[2] += a0 * c2;
        h[3] += a1 * c0; h[4] += a1 * c1; h[5] += a1 * c2;
        h[6] += a2 * c0; h[7] += a2 * c1; h[8] += a2 * c2;
    }
    float Hm[9];
#pragma unroll
    for (int k = 0; k < 9; ++k) Hm[k] = block_red(h[k], red, OP_ADD);

    // ---- thread 0: 3x3 SVD (Jacobi on H^T H) + Kabsch assembly ----
    if (tid == 0) {
        float S[9];
        for (int i = 0; i < 3; ++i)
            for (int j = 0; j < 3; ++j) {
                float a = 0.f;
                for (int k = 0; k < 3; ++k) a += Hm[k * 3 + i] * Hm[k * 3 + j];
                S[i * 3 + j] = a;
            }
        float V[9] = {1, 0, 0, 0, 1, 0, 0, 0, 1};
        for (int sweep = 0; sweep < 12; ++sweep) {
            for (int pi = 0; pi < 3; ++pi) {
                const int p = (pi == 2) ? 1 : 0;
                const int q = (pi == 0) ? 1 : 2;
                float apq = S[p * 3 + q];
                float app = S[p * 3 + p], aqq = S[q * 3 + q];
                if (fabsf(apq) <= 1e-12f * (fabsf(app) + fabsf(aqq)) + 1e-30f) continue;
                float tau = (aqq - app) / (2.f * apq);
                float tt  = (tau >= 0.f ? 1.f : -1.f) / (fabsf(tau) + sqrtf(1.f + tau * tau));
                float c   = 1.f / sqrtf(1.f + tt * tt), s = tt * c;
                for (int k = 0; k < 3; ++k) {
                    float skp = S[k * 3 + p], skq = S[k * 3 + q];
                    S[k * 3 + p] = c * skp - s * skq; S[k * 3 + q] = s * skp + c * skq;
                }
                for (int k = 0; k < 3; ++k) {
                    float spk = S[p * 3 + k], sqk = S[q * 3 + k];
                    S[p * 3 + k] = c * spk - s * sqk; S[q * 3 + k] = s * spk + c * sqk;
                }
                for (int k = 0; k < 3; ++k) {
                    float vkp = V[k * 3 + p], vkq = V[k * 3 + q];
                    V[k * 3 + p] = c * vkp - s * vkq; V[k * 3 + q] = s * vkp + c * vkq;
                }
            }
        }
        // sort eigenvalues descending (swap V columns accordingly)
        float ev[3] = {S[0], S[4], S[8]};
        for (int a = 0; a < 2; ++a)
            for (int c2 = 0; c2 < 2 - a; ++c2)
                if (ev[c2] < ev[c2 + 1]) {
                    float t = ev[c2]; ev[c2] = ev[c2 + 1]; ev[c2 + 1] = t;
                    for (int k = 0; k < 3; ++k) {
                        float t2 = V[k * 3 + c2]; V[k * 3 + c2] = V[k * 3 + c2 + 1]; V[k * 3 + c2 + 1] = t2;
                    }
                }
        // U columns from H*v_k, Gram-Schmidt, right-handed completion
        float hv[3][3];
        for (int k = 0; k < 3; ++k)
            for (int i = 0; i < 3; ++i) {
                float a = 0.f;
                for (int j = 0; j < 3; ++j) a += Hm[i * 3 + j] * V[j * 3 + k];
                hv[k][i] = a;
            }
        float u0[3], u1[3];
        float n0v = sqrtf(hv[0][0] * hv[0][0] + hv[0][1] * hv[0][1] + hv[0][2] * hv[0][2]);
        if (n0v > 1e-20f) { u0[0] = hv[0][0] / n0v; u0[1] = hv[0][1] / n0v; u0[2] = hv[0][2] / n0v; }
        else { u0[0] = 1.f; u0[1] = 0.f; u0[2] = 0.f; }
        float d01 = u0[0] * hv[1][0] + u0[1] * hv[1][1] + u0[2] * hv[1][2];
        u1[0] = hv[1][0] - d01 * u0[0]; u1[1] = hv[1][1] - d01 * u0[1]; u1[2] = hv[1][2] - d01 * u0[2];
        float n1v = sqrtf(u1[0] * u1[0] + u1[1] * u1[1] + u1[2] * u1[2]);
        if (n1v > 1e-20f) { u1[0] /= n1v; u1[1] /= n1v; u1[2] /= n1v; }
        else {
            float ax = (fabsf(u0[0]) < 0.9f) ? 1.f : 0.f;
            float t3[3] = {ax, 1.f - ax, 0.f};
            float dd = t3[0] * u0[0] + t3[1] * u0[1] + t3[2] * u0[2];
            u1[0] = t3[0] - dd * u0[0]; u1[1] = t3[1] - dd * u0[1]; u1[2] = t3[2] - dd * u0[2];
            float nn = sqrtf(u1[0] * u1[0] + u1[1] * u1[1] + u1[2] * u1[2]) + 1e-30f;
            u1[0] /= nn; u1[1] /= nn; u1[2] /= nn;
        }
        float u2[3] = {u0[1] * u1[2] - u0[2] * u1[1],
                       u0[2] * u1[0] - u0[0] * u1[2],
                       u0[0] * u1[1] - u0[1] * u1[0]};
        // flip v2 so that H v2 aligns with u2 (keeps sigma_2 >= 0 consistent)
        float d2 = u2[0] * hv[2][0] + u2[1] * hv[2][1] + u2[2] * hv[2][2];
        if (d2 < 0.f) { V[2] = -V[2]; V[5] = -V[5]; V[8] = -V[8]; }
        float U[9];
        for (int i = 0; i < 3; ++i) { U[i * 3 + 0] = u0[i]; U[i * 3 + 1] = u1[i]; U[i * 3 + 2] = u2[i]; }
        // r = V U^T ; det; reflection fix on V's 3rd column; R = V_adj U^T
        float r[9];
        for (int i = 0; i < 3; ++i)
            for (int j = 0; j < 3; ++j) {
                float a = 0.f;
                for (int k = 0; k < 3; ++k) a += V[i * 3 + k] * U[j * 3 + k];
                r[i * 3 + j] = a;
            }
        float det = r[0] * (r[4] * r[8] - r[5] * r[7])
                  - r[1] * (r[3] * r[8] - r[5] * r[6])
                  + r[2] * (r[3] * r[7] - r[4] * r[6]);
        float sgn = (det < 0.f) ? -1.f : 1.f;
        V[2] *= sgn; V[5] *= sgn; V[8] *= sgn;
        float R[9];
        for (int i = 0; i < 3; ++i)
            for (int j = 0; j < 3; ++j) {
                float a = 0.f;
                for (int k = 0; k < 3; ++k) a += V[i * 3 + k] * U[j * 3 + k];
                R[i * 3 + j] = a;
            }
        float t0 = -(R[0] * sm0 + R[1] * sm1 + R[2] * sm2) + cm0;
        float t1 = -(R[3] * sm0 + R[4] * sm1 + R[5] * sm2) + cm1;
        float t2 = -(R[6] * sm0 + R[7] * sm1 + R[8] * sm2) + cm2;
        for (int k = 0; k < 9; ++k) out[b * 9 + k] = R[k];
        out[Bb * 9 + b * 3 + 0] = t0;
        out[Bb * 9 + b * 3 + 1] = t1;
        out[Bb * 9 + b * 3 + 2] = t2;
    }
}

// =====================================================================
extern "C" void kernel_launch(void* const* d_in, const int* in_sizes, int n_in,
                              void* d_out, int out_size, void* d_ws, size_t ws_size,
                              hipStream_t stream)
{
    const float* src_emb = (const float*)d_in[0];
    const float* tgt_emb = (const float*)d_in[1];
    const float* src     = (const float*)d_in[2];
    const float* tgt     = (const float*)d_in[3];
    const float* src_dm  = (const float*)d_in[4];
    const float* tgt_dm  = (const float*)d_in[5];
    const float* src_im  = (const float*)d_in[6];
    const float* tgt_im  = (const float*)d_in[7];

    float* ws       = (float*)d_ws;
    float* ws_corr  = ws;                         // B*3*N floats
    float* ws_mval  = ws + (size_t)Bb * 3 * Nn;   // B*N floats

    const size_t shmem = (size_t)(ROWS * AS + MT * BS) * 2 + 5 * MT * 4;

    flash_corr_kernel<<<dim3(Bb * (Nn / ROWS)), dim3(THREADS), shmem, stream>>>(
        src_emb, tgt_emb, tgt, tgt_dm, tgt_im, src_dm, src_im, ws_corr, ws_mval);

    reduce_svd_kernel<<<dim3(Bb), dim3(256), 0, stream>>>(
        src, ws_corr, ws_mval, (float*)d_out);
}
